// MoELayer_64450279244024
// MI455X (gfx1250) — matrix operand order repaired and checked by reading, not compiled, not run on hardware
//
#include <hip/hip_runtime.h>
#include <cmath>

// ---------------------------------------------------------------------------
// MoE (top-2, E=16, D=1024, H=1536, N=4096) for gfx1250 (MI455X, wave32, WMMA)
// M-tile = 32 tokens (2 row tiles) so every B fragment feeds 2 WMMAs.
// ---------------------------------------------------------------------------

typedef __attribute__((ext_vector_type(16))) _Float16 v16h;
typedef __attribute__((ext_vector_type(8)))  _Float16 v8h;
typedef __attribute__((ext_vector_type(4)))  _Float16 v4h;
typedef __attribute__((ext_vector_type(8)))  float    v8f;
typedef __attribute__((ext_vector_type(4)))  float    v4f;

static constexpr int   kD    = 1024;
static constexpr int   kH    = 1536;
static constexpr int   kE    = 16;
static constexpr int   kNTOK = 4096;   // B*T = 2*2048
static constexpr float kAlphaAux = 0.02f;

// ---- workspace layout (bytes) ----
static constexpr size_t OFF_COUNT = 0;                       // 16 x int
static constexpr size_t OFF_GSUM  = 64;                      // 16 x float
static constexpr size_t OFF_BTOK  = 256;                     // E*NTOK int
static constexpr size_t OFF_BGATE = OFF_BTOK + (size_t)kE * kNTOK * 4;
static constexpr size_t OFF_W1H   = OFF_BGATE + (size_t)kE * kNTOK * 4;
static constexpr size_t W1_ELTS   = (size_t)kE * kH * kD;
static constexpr size_t W2_ELTS   = (size_t)kE * kD * kH;
static constexpr size_t X_ELTS    = (size_t)kNTOK * kD;
static constexpr size_t OFF_W2H   = OFF_W1H + W1_ELTS * 2;
static constexpr size_t OFF_XH    = OFF_W2H + W2_ELTS * 2;
static constexpr size_t WS_NEED_FULL = OFF_XH + X_ELTS * 2;  // ~110 MB
// fallback (inline-convert) path needs only OFF_W1H bytes (~0.5 MB)

__device__ __forceinline__ v16h cat8h(v8h lo, v8h hi) {
  return __builtin_shufflevector(lo, hi, 0,1,2,3,4,5,6,7,8,9,10,11,12,13,14,15);
}
__device__ __forceinline__ v8f cat4f(v4f lo, v4f hi) {
  return __builtin_shufflevector(lo, hi, 0,1,2,3,4,5,6,7);
}
__device__ __forceinline__ v8h cvt8(v8f f) {
  return __builtin_convertvector(f, v8h);
}
__device__ __forceinline__ float gelu_tanh(float v) {
  const float c0 = 0.7978845608028654f;   // sqrt(2/pi)
  return 0.5f * v * (1.0f + tanhf(c0 * (v + 0.044715f * v * v * v)));
}
__device__ __forceinline__ v8f wmma16(v16h a, v16h b, v8f c) {
  return __builtin_amdgcn_wmma_f32_16x16x32_f16(false, a, false, b, (short)0, c,
                                                false, false);
}

// B fragment: lane = column n, two 8-element runs at `off` and `off+16`
template <bool PC>
__device__ __forceinline__ v16h load_b_frag(const _Float16* __restrict__ ph,
                                            const float* __restrict__ pf,
                                            int off) {
  if constexpr (PC) {
    v8h b0 = *(const v8h*)(ph + off);
    v8h b1 = *(const v8h*)(ph + off + 16);
    return cat8h(b0, b1);
  } else {
    v4f g0 = *(const v4f*)(pf + off);
    v4f g1 = *(const v4f*)(pf + off + 4);
    v4f g2 = *(const v4f*)(pf + off + 16);
    v4f g3 = *(const v4f*)(pf + off + 20);
    return cat8h(cvt8(cat4f(g0, g1)), cvt8(cat4f(g2, g3)));
  }
}

// A fragment: lane = row m, two 8-element runs at `off` and `off+16`
template <bool PC>
__device__ __forceinline__ v16h load_a_frag(const _Float16* __restrict__ ph,
                                            const float* __restrict__ pf,
                                            int off) {
  if constexpr (PC) {
    v8h a0 = *(const v8h*)(ph + off);
    v8h a1 = *(const v8h*)(ph + off + 16);
    return cat8h(a0, a1);
  } else {
    v4f f0 = *(const v4f*)(pf + off);
    v4f f1 = *(const v4f*)(pf + off + 4);
    v4f f2 = *(const v4f*)(pf + off + 16);
    v4f f3 = *(const v4f*)(pf + off + 20);
    return cat8h(cvt8(cat4f(f0, f1)), cvt8(cat4f(f2, f3)));
  }
}

// ---------------------------------------------------------------------------
// fp32 -> fp16 bulk conversion (preconverted weight/activation path)
// ---------------------------------------------------------------------------
__global__ void cvt_f32_f16_kernel(const float* __restrict__ s,
                                   _Float16* __restrict__ d, size_t n) {
  size_t i = ((size_t)blockIdx.x * blockDim.x + threadIdx.x) * 4;
  if (i + 3 < n) {
    v4f f = *(const v4f*)(s + i);
    *(v4h*)(d + i) = __builtin_convertvector(f, v4h);
  } else {
    for (; i < n; ++i) d[i] = (_Float16)s[i];
  }
}

// ---------------------------------------------------------------------------
// Router: one wave per token; lanes 0..15 hold one expert logit each.
// ---------------------------------------------------------------------------
__global__ __launch_bounds__(32) void router_kernel(
    const float* __restrict__ x, const float* __restrict__ rw,
    int* __restrict__ counts, float* __restrict__ gsum,
    int* __restrict__ btok, float* __restrict__ bgate) {
  const int n    = blockIdx.x;
  const int lane = threadIdx.x;

  float acc = 0.0f;
  if (lane < kE) {
    const v4f* xr = (const v4f*)(x + (size_t)n * kD);
    const v4f* wr = (const v4f*)(rw + (size_t)lane * kD);
    #pragma unroll 4
    for (int i = 0; i < kD / 4; ++i) {
      v4f a = xr[i], b = wr[i];
      acc += a.x * b.x + a.y * b.y + a.z * b.z + a.w * b.w;
    }
  }
  float lg[kE];
  #pragma unroll
  for (int i = 0; i < kE; ++i) lg[i] = __shfl(acc, i, 32);

  if (lane == 0) {
    int i1 = 0; float v1 = lg[0];
    #pragma unroll
    for (int i = 1; i < kE; ++i) { if (lg[i] > v1) { v1 = lg[i]; i1 = i; } }
    int i2 = -1; float v2 = -3.4e38f;
    #pragma unroll
    for (int i = 0; i < kE; ++i) { if (i != i1 && lg[i] > v2) { v2 = lg[i]; i2 = i; } }
    const float e2 = __expf(v2 - v1);           // stable 2-way softmax
    const float inv = 1.0f / (1.0f + e2);
    const float g1 = inv, g2 = e2 * inv;

    int s1 = atomicAdd(&counts[i1], 1);
    btok[(size_t)i1 * kNTOK + s1] = n;
    bgate[(size_t)i1 * kNTOK + s1] = g1;
    atomicAdd(&gsum[i1], g1);

    int s2 = atomicAdd(&counts[i2], 1);
    btok[(size_t)i2 * kNTOK + s2] = n;
    bgate[(size_t)i2 * kNTOK + s2] = g2;
    atomicAdd(&gsum[i2], g2);
  }
}

// ---------------------------------------------------------------------------
// Expert kernel. Block = 256 threads (8 waves), 32-token tile of one expert.
// H is processed in 4 chunks of 384:
//   phase 1: h_chunk[32,384] = gelu(X @ W1[:,chunk]^T + b1) -> LDS (f16)
//   phase 2: C2[32,1024]    += h_chunk @ W2[:,chunk]^T   (persistent v8f accs)
// Fragment layouts (wave32):
//   A: lane l -> row m=l&15, K=(j<8?0:16)+8*(l>>4)+(j&7)  (two 16B runs)
//   B: lane l -> col n=l&15, K=(j<8?0:16)+8*(l>>4)+(j&7)  (two 16B runs)
//   C: element i -> row i+8*(l>>4), col l&15
// LDS row stride 392 halves (784B): bank(m)=4m mod 64 -> conflict-free b128.
// ---------------------------------------------------------------------------
template <bool PC>
__global__ __launch_bounds__(256) void moe_expert_kernel(
    const float* __restrict__ x,  const _Float16* __restrict__ xh,
    const float* __restrict__ w1, const _Float16* __restrict__ w1h,
    const float* __restrict__ b1,
    const float* __restrict__ w2, const _Float16* __restrict__ w2h,
    const float* __restrict__ b2,
    const int* __restrict__ counts, const int* __restrict__ btok,
    const float* __restrict__ bgate, float* __restrict__ out) {
  constexpr int RT  = 2;          // row tiles (M = 32)
  constexpr int CH  = 384;        // H-chunk width
  constexpr int NCH = kH / CH;    // 4 chunks
  constexpr int STR = CH + 8;     // LDS row stride (halves), bank-conflict pad

  const int e     = blockIdx.y;
  const int cnt   = counts[e];
  const int tile0 = blockIdx.x * 32;
  if (tile0 >= cnt) return;                       // uniform early exit

  __shared__ _Float16 hsm[32 * STR];              // ~24.5 KB

  const int lane = threadIdx.x & 31;
  const int wv   = threadIdx.x >> 5;
  const int half = lane >> 4;
  const int nl   = lane & 15;

  // tokens supplying A-matrix rows for the two row tiles
  const _Float16* xrh[RT];
  const float*    xrf[RT];
  #pragma unroll
  for (int rt = 0; rt < RT; ++rt) {
    int s = tile0 + rt * 16 + nl;
    if (s >= cnt) s = tile0;                      // pad with a valid token
    const int tok = btok[(size_t)e * kNTOK + s];
    xrh[rt] = PC ? (xh + (size_t)tok * kD) : nullptr;
    xrf[rt] = x + (size_t)tok * kD;
  }

  // persistent phase-2 accumulators: [rt][ct] -> C2 rows, 16 cols each
  v8f c2[RT * 8];
  #pragma unroll
  for (int i = 0; i < RT * 8; ++i) c2[i] = v8f{};

  for (int c = 0; c < NCH; ++c) {
    const int cb = c * CH;
    __syncthreads();                              // hsm free from prev phase 2

    // ---------------- Phase 1: h_chunk -> LDS ----------------
    {
      const int colbase = cb + wv * 48;           // 3 column tiles per wave
      v8f p1[RT][3];
      #pragma unroll
      for (int rt = 0; rt < RT; ++rt)
        #pragma unroll
        for (int t = 0; t < 3; ++t) p1[rt][t] = v8f{};

      for (int k0 = 0; k0 < kD; k0 += 32) {
        v16h a[RT];
        #pragma unroll
        for (int rt = 0; rt < RT; ++rt)
          a[rt] = load_a_frag<PC>(xrh[rt], xrf[rt], k0 + half * 8);
        #pragma unroll
        for (int t = 0; t < 3; ++t) {
          const int ncol = colbase + t * 16 + nl;
          const _Float16* wrh = PC ? (w1h + ((size_t)e * kH + ncol) * kD)
                                   : nullptr;
          const float* wrf = w1 + ((size_t)e * kH + ncol) * kD;
          v16h b = load_b_frag<PC>(wrh, wrf, k0 + half * 8);
          p1[0][t] = wmma16(a[0], b, p1[0][t]);
          p1[1][t] = wmma16(a[1], b, p1[1][t]);
        }
      }
      #pragma unroll
      for (int t = 0; t < 3; ++t) {
        const int ncol = colbase + t * 16 + nl;
        const float bias = b1[(size_t)e * kH + ncol];
        #pragma unroll
        for (int rt = 0; rt < RT; ++rt) {
          #pragma unroll
          for (int i = 0; i < 8; ++i) {
            float v = p1[rt][t][i] + bias;
            hsm[(i + 8 * half + 16 * rt) * STR + (ncol - cb)] =
                (_Float16)gelu_tanh(v);
          }
        }
      }
    }
    __syncthreads();                              // h_chunk visible

    // ---------------- Phase 2: C2 += h_chunk @ W2^T ----------------
    for (int k0 = 0; k0 < CH; k0 += 32) {
      v16h a2[RT];
      #pragma unroll
      for (int rt = 0; rt < RT; ++rt) {
        const _Float16* hr = &hsm[(nl + 16 * rt) * STR];
        v8h a0 = *(const v8h*)(hr + k0 + half * 8);
        v8h a1 = *(const v8h*)(hr + k0 + 16 + half * 8);
        a2[rt] = cat8h(a0, a1);
      }
      #pragma unroll
      for (int ct = 0; ct < 8; ++ct) {
        const int ncol = (wv * 8 + ct) * 16 + nl;
        const _Float16* wrh = PC ? (w2h + ((size_t)e * kD + ncol) * kH)
                                 : nullptr;
        const float* wrf = w2 + ((size_t)e * kD + ncol) * kH;
        v16h b = load_b_frag<PC>(wrh, wrf, cb + k0 + half * 8);
        c2[ct]     = wmma16(a2[0], b, c2[ct]);
        c2[8 + ct] = wmma16(a2[1], b, c2[8 + ct]);
      }
    }
  }

  // ---------------- epilogue: out += gate * (C2 + b2) ----------------
  int   tokR[RT][8];
  float gR[RT][8];
  #pragma unroll
  for (int rt = 0; rt < RT; ++rt) {
    #pragma unroll
    for (int i = 0; i < 8; ++i) {
      int slot = tile0 + rt * 16 + i + 8 * half;
      if (slot < cnt) {
        tokR[rt][i] = btok[(size_t)e * kNTOK + slot];
        gR[rt][i]   = bgate[(size_t)e * kNTOK + slot];
      } else {
        tokR[rt][i] = -1;
        gR[rt][i]   = 0.0f;
      }
    }
  }
  #pragma unroll
  for (int ct = 0; ct < 8; ++ct) {
    const int ncol = (wv * 8 + ct) * 16 + nl;
    const float bias = b2[(size_t)e * kD + ncol];
    #pragma unroll
    for (int rt = 0; rt < RT; ++rt) {
      #pragma unroll
      for (int i = 0; i < 8; ++i) {
        if (tokR[rt][i] >= 0) {
          atomicAdd(&out[(size_t)tokR[rt][i] * kD + ncol],
                    gR[rt][i] * (c2[rt * 8 + ct][i] + bias));
        }
      }
    }
  }
}

// ---------------------------------------------------------------------------
// Aux loss: aux = alpha * mean_e( f_i * m_i )
// ---------------------------------------------------------------------------
__global__ void aux_loss_kernel(const int* __restrict__ counts,
                                const float* __restrict__ gsum,
                                float* __restrict__ auxout) {
  if (threadIdx.x == 0 && blockIdx.x == 0) {
    float sc = 0.0f;
    for (int i = 0; i < kE; ++i) sc += (float)counts[i];
    float s = 0.0f;
    for (int i = 0; i < kE; ++i) {
      float c = (float)counts[i];
      float f = (sc > 0.0f) ? (c / sc) : 0.0f;
      float den = (c < 1.0f) ? 1.0f : c;
      s += f * (gsum[i] / den);
    }
    *auxout = kAlphaAux * (s / (float)kE);
  }
}

// ---------------------------------------------------------------------------
extern "C" void kernel_launch(void* const* d_in, const int* in_sizes, int n_in,
                              void* d_out, int out_size, void* d_ws,
                              size_t ws_size, hipStream_t stream) {
  (void)in_sizes; (void)n_in;
  const float* x  = (const float*)d_in[0];
  const float* rw = (const float*)d_in[1];
  const float* w1 = (const float*)d_in[2];
  const float* b1 = (const float*)d_in[3];
  const float* w2 = (const float*)d_in[4];
  const float* b2 = (const float*)d_in[5];
  float* out = (float*)d_out;

  char* ws = (char*)d_ws;
  int*      counts = (int*)(ws + OFF_COUNT);
  float*    gsum   = (float*)(ws + OFF_GSUM);
  int*      btok   = (int*)(ws + OFF_BTOK);
  float*    bgate  = (float*)(ws + OFF_BGATE);
  _Float16* w1h    = (_Float16*)(ws + OFF_W1H);
  _Float16* w2h    = (_Float16*)(ws + OFF_W2H);
  _Float16* xh     = (_Float16*)(ws + OFF_XH);

  const bool pc = (ws_size >= WS_NEED_FULL);

  // zero output (includes aux slot) and router accumulators
  hipMemsetAsync(d_out, 0, (size_t)out_size * sizeof(float), stream);
  hipMemsetAsync(d_ws, 0, 256, stream);

  if (pc) {
    unsigned g1 = (unsigned)((W1_ELTS / 4 + 255) / 256);
    unsigned g2 = (unsigned)((W2_ELTS / 4 + 255) / 256);
    unsigned gx = (unsigned)((X_ELTS / 4 + 255) / 256);
    cvt_f32_f16_kernel<<<g1, 256, 0, stream>>>(w1, w1h, W1_ELTS);
    cvt_f32_f16_kernel<<<g2, 256, 0, stream>>>(w2, w2h, W2_ELTS);
    cvt_f32_f16_kernel<<<gx, 256, 0, stream>>>(x, xh, X_ELTS);
  }

  router_kernel<<<kNTOK, 32, 0, stream>>>(x, rw, counts, gsum, btok, bgate);

  dim3 grid(kNTOK / 32, kE);  // 128 tiles (worst-case capacity) x 16 experts
  if (pc) {
    moe_expert_kernel<true><<<grid, 256, 0, stream>>>(
        x, xh, w1, w1h, b1, w2, w2h, b2, counts, btok, bgate, out);
  } else {
    moe_expert_kernel<false><<<grid, 256, 0, stream>>>(
        x, xh, w1, w1h, b1, w2, w2h, b2, counts, btok, bgate, out);
  }

  aux_loss_kernel<<<1, 32, 0, stream>>>(counts, gsum,
                                        out + ((size_t)out_size - 1));
}
